// GNNWithDynamicK_45672682226325
// MI455X (gfx1250) — matrix operand order repaired and checked by reading, not compiled, not run on hardware
//
#include <hip/hip_runtime.h>
#include <hip/hip_bf16.h>

typedef __attribute__((ext_vector_type(16))) _Float16 v16h;
typedef __attribute__((ext_vector_type(2)))  _Float16 v2h;
typedef __attribute__((ext_vector_type(8)))  float    v8f;

#define N_NODES 50000
#define N_EDGES 1200000
#define N_OD    10000
#define HID     64

// ---------------------------------------------------------------------------
// A-fragment swizzle for V_WMMA_F32_16X16X32_F16 (16-bit A 16x32, ISA 7.12.2):
//   lanes 0-15 : M=0..15, V0..3 <-> K{0,1},{2,3},{4,5},{6,7}, V4..7 <-> K16..23
//   lanes16-31 : M=0..15, V0..3 <-> K{8,9}..{14,15},          V4..7 <-> K24..31
// Packed variant: K pair (k0, k0+1), k0 even, lands in one 32-bit half-pair
// of one lane's fragment -> single ds_store_b32.
// ---------------------------------------------------------------------------
__device__ __forceinline__ void storeA2(v16h (*af)[32], int m, int k0,
                                        float a, float b) {
    int ks = k0 >> 5;
    int kk = k0 & 31;                       // even
    int vgBase = (kk >= 16) ? 4 : 0;
    int k2 = kk & 15;
    int kgroup = k2 >> 3;                   // 0 -> lanes 0-15, 1 -> lanes 16-31
    int vg = vgBase + ((k2 & 7) >> 1);
    int e  = 2 * vg;                        // even element -> 4B aligned
    int ln = kgroup * 16 + m;
    *(v2h*)((_Float16*)&af[ks][ln] + e) = (v2h){(_Float16)a, (_Float16)b};
}

// ---------------------------------------------------------------------------
// h = relu(nf @ Wn + bn)   [50000,16]@[16,64]
// ---------------------------------------------------------------------------
__global__ void __launch_bounds__(256) nodeproj_kernel(
    const float* __restrict__ nf, const float* __restrict__ Wn,
    const float* __restrict__ bn, float* __restrict__ h, int n)
{
    int idx = blockIdx.x * 256 + threadIdx.x;
    if (idx >= n * HID) return;
    int node = idx >> 6, c = idx & 63;
    const float* r = nf + (size_t)node * 16;
    float acc = bn[c];
#pragma unroll
    for (int k = 0; k < 16; ++k) acc += r[k] * Wn[k * HID + c];
    h[idx] = fmaxf(acc, 0.f);
}

__global__ void __launch_bounds__(256) zero_kernel(float* __restrict__ p, int n)
{
    int stride = gridDim.x * 256;
    for (int i = blockIdx.x * 256 + threadIdx.x; i < n; i += stride) p[i] = 0.f;
}

// ---------------------------------------------------------------------------
// Stage a [128,64] weight matrix pre-swizzled into WMMA B fragments.
// B 32x16 per step: lanes0-15 K=0..7,16..23, lanes16-31 K=8..15,24..31 (A order),
// N = lane&15  (ISA 7.12.4 B layout, K=32 dense case).
// ---------------------------------------------------------------------------
__device__ __forceinline__ void stageB(v16h (*wfrag)[4][32],
                                       const float* __restrict__ W, int tid) {
    for (int i = tid; i < 4 * 4 * 32 * 16; i += 256) {
        int e  = i & 15;
        int ln = (i >> 4) & 31;
        int ks = (i >> 9) & 3;
        int nt = (i >> 11) & 3;
        int nn = nt * 16 + (ln & 15);
        int kk = ks * 32 + ((e >> 3) * 16) + ((ln >> 4) * 8) + (e & 7);
        ((_Float16*)&wfrag[nt][ks][ln])[e] = (_Float16)W[kk * HID + nn];
    }
}

// ---------------------------------------------------------------------------
// Message kernel (dominant): per 16-edge tile
//   x = [h[src] | relu(ef@We+be)]  (16x128), m = relu(x @ Wm + bm) (16x64)
//   atomic scatter-add m into agg[dst].
// 8 waves/block, one tile per wave, grid-stride over 128-edge groups.
// ---------------------------------------------------------------------------
__global__ void __launch_bounds__(256) msg_kernel(
    const float* __restrict__ h,  const float* __restrict__ ef,
    const int*  __restrict__ src, const int*  __restrict__ dst,
    const float* __restrict__ We, const float* __restrict__ be,
    const float* __restrict__ Wm, const float* __restrict__ bm,
    float* __restrict__ agg, int nGroups)
{
    __shared__ v16h  wfrag[4][4][32];     // [ntile][kstep][lane]  16 KB
    __shared__ v16h  afrag[8][4][32];     // [wave][kstep][lane]   32 KB
    __shared__ float efTile[8][16][8];    // 4 KB
    __shared__ int   dstIdx[8][16];
    __shared__ int   srcIdx[8][16];
    __shared__ float sWe[8 * HID];
    __shared__ float sbe[HID];
    __shared__ float sbm[HID];

    const int tid  = threadIdx.x;
    const int lane = tid & 31;
    const int wave = tid >> 5;

    for (int i = tid; i < 8 * HID; i += 256) sWe[i] = We[i];
    for (int i = tid; i < HID; i += 256) { sbe[i] = be[i]; sbm[i] = bm[i]; }
    stageB(wfrag, Wm, tid);
    __syncthreads();

    const int kgroup = lane >> 4;
    const int nl     = lane & 15;

    // hoisted per-lane constants
    float bv[4];
#pragma unroll
    for (int nt = 0; nt < 4; ++nt) bv[nt] = sbm[nt * 16 + nl];
    float wcol[2][8];
    float bc0 = sbe[2 * lane], bc1 = sbe[2 * lane + 1];
#pragma unroll
    for (int j = 0; j < 8; ++j) {
        wcol[0][j] = sWe[j * HID + 2 * lane];
        wcol[1][j] = sWe[j * HID + 2 * lane + 1];
    }

    for (int g = blockIdx.x; g < nGroups; g += gridDim.x) {
        const int e0 = g * 128 + wave * 16;

        if (lane < 16) {
            dstIdx[wave][lane] = dst[e0 + lane];
            srcIdx[wave][lane] = src[e0 + lane];
        }
        // edge-feature tile: 16x8 f32, one float4 per lane
        ((float4*)&efTile[wave][0][0])[lane] =
            ((const float4*)(ef + (size_t)e0 * 8))[lane];
        __syncthreads();

        // gather h[src] rows -> A cols 0..63 (float2 per lane, packed f16x2 store)
        for (int m = 0; m < 16; ++m) {
            const float2 v = ((const float2*)(h + (size_t)srcIdx[wave][m] * HID))[lane];
            storeA2(afrag[wave], m, 2 * lane, v.x, v.y);
        }
        // fused edge projection -> A cols 64..127 (channels 2*lane, 2*lane+1)
        for (int m = 0; m < 16; ++m) {
            float a0 = bc0, a1 = bc1;
#pragma unroll
            for (int j = 0; j < 8; ++j) {
                float x = efTile[wave][m][j];
                a0 += x * wcol[0][j];
                a1 += x * wcol[1][j];
            }
            storeA2(afrag[wave], m, 64 + 2 * lane, fmaxf(a0, 0.f), fmaxf(a1, 0.f));
        }
        __syncthreads();

        v16h a0 = afrag[wave][0][lane];
        v16h a1 = afrag[wave][1][lane];
        v16h a2 = afrag[wave][2][lane];
        v16h a3 = afrag[wave][3][lane];

        // 4 independent accumulators, interleaved across K-steps (no RAW chain)
        v8f acc[4] = {{}, {}, {}, {}};
#pragma unroll
        for (int nt = 0; nt < 4; ++nt)
            acc[nt] = __builtin_amdgcn_wmma_f32_16x16x32_f16(false, a0, false, wfrag[nt][0][lane], (short)0, acc[nt], false, false);
#pragma unroll
        for (int nt = 0; nt < 4; ++nt)
            acc[nt] = __builtin_amdgcn_wmma_f32_16x16x32_f16(false, a1, false, wfrag[nt][1][lane], (short)0, acc[nt], false, false);
#pragma unroll
        for (int nt = 0; nt < 4; ++nt)
            acc[nt] = __builtin_amdgcn_wmma_f32_16x16x32_f16(false, a2, false, wfrag[nt][2][lane], (short)0, acc[nt], false, false);
#pragma unroll
        for (int nt = 0; nt < 4; ++nt)
            acc[nt] = __builtin_amdgcn_wmma_f32_16x16x32_f16(false, a3, false, wfrag[nt][3][lane], (short)0, acc[nt], false, false);

        // scatter: one 64-bit row base per M-row; N-tile steps fold into the
        // 24-bit instruction offset of global_atomic_add_f32
#pragma unroll
        for (int r = 0; r < 8; ++r) {                 // C/D: VGPR r -> M = r + 8*kgroup
            int Mrow = r + kgroup * 8;
            float* base = agg + (size_t)dstIdx[wave][Mrow] * HID + nl;
#pragma unroll
            for (int nt = 0; nt < 4; ++nt)
                unsafeAtomicAdd(base + nt * 16, fmaxf(acc[nt][r] + bv[nt], 0.f));
        }
        __syncthreads();
    }
}

// ---------------------------------------------------------------------------
// Update kernel: h = relu([h | agg] @ Wu + bu) + h, in place (per-node tiles).
// ---------------------------------------------------------------------------
__global__ void __launch_bounds__(256) upd_kernel(
    float* __restrict__ h, const float* __restrict__ agg,
    const float* __restrict__ Wu, const float* __restrict__ bu,
    int nTiles, int nGroups)
{
    __shared__ v16h  wfrag[4][4][32];
    __shared__ v16h  afrag[8][4][32];
    __shared__ float sbu[HID];

    const int tid  = threadIdx.x;
    const int lane = tid & 31;
    const int wave = tid >> 5;

    for (int i = tid; i < HID; i += 256) sbu[i] = bu[i];
    stageB(wfrag, Wu, tid);
    __syncthreads();

    const int kgroup = lane >> 4;
    const int nl     = lane & 15;
    float bv[4];
#pragma unroll
    for (int nt = 0; nt < 4; ++nt) bv[nt] = sbu[nt * 16 + nl];

    for (int g = blockIdx.x; g < nGroups; g += gridDim.x) {
        const int tile  = g * 8 + wave;
        const bool ok   = tile < nTiles;
        const int node0 = tile * 16;

        if (ok) {
            for (int m = 0; m < 16; ++m) {
                const float2 vh = ((const float2*)(h   + (size_t)(node0 + m) * HID))[lane];
                const float2 va = ((const float2*)(agg + (size_t)(node0 + m) * HID))[lane];
                storeA2(afrag[wave], m,      2 * lane, vh.x, vh.y);
                storeA2(afrag[wave], m, 64 + 2 * lane, va.x, va.y);
            }
        }
        __syncthreads();
        if (ok) {
            v16h a0 = afrag[wave][0][lane];
            v16h a1 = afrag[wave][1][lane];
            v16h a2 = afrag[wave][2][lane];
            v16h a3 = afrag[wave][3][lane];
            v8f acc[4] = {{}, {}, {}, {}};
#pragma unroll
            for (int nt = 0; nt < 4; ++nt)
                acc[nt] = __builtin_amdgcn_wmma_f32_16x16x32_f16(false, a0, false, wfrag[nt][0][lane], (short)0, acc[nt], false, false);
#pragma unroll
            for (int nt = 0; nt < 4; ++nt)
                acc[nt] = __builtin_amdgcn_wmma_f32_16x16x32_f16(false, a1, false, wfrag[nt][1][lane], (short)0, acc[nt], false, false);
#pragma unroll
            for (int nt = 0; nt < 4; ++nt)
                acc[nt] = __builtin_amdgcn_wmma_f32_16x16x32_f16(false, a2, false, wfrag[nt][2][lane], (short)0, acc[nt], false, false);
#pragma unroll
            for (int nt = 0; nt < 4; ++nt)
                acc[nt] = __builtin_amdgcn_wmma_f32_16x16x32_f16(false, a3, false, wfrag[nt][3][lane], (short)0, acc[nt], false, false);

#pragma unroll
            for (int r = 0; r < 8; ++r) {
                int node = node0 + r + kgroup * 8;
                float* base = h + (size_t)node * HID + nl;
#pragma unroll
                for (int nt = 0; nt < 4; ++nt) {
                    float* p = base + nt * 16;
                    *p = fmaxf(acc[nt][r] + bv[nt], 0.f) + *p;   // residual, in place
                }
            }
        }
        __syncthreads();
    }
}

// ---------------------------------------------------------------------------
// scores = relu([h[a]|h[b]] @ Ws1 + bs1) @ Ws2 + bs2   (one wave per OD pair)
// ---------------------------------------------------------------------------
__global__ void __launch_bounds__(256) score_kernel(
    const float* __restrict__ h, const int* __restrict__ od,
    const float* __restrict__ Ws1, const float* __restrict__ bs1,
    const float* __restrict__ Ws2, const float* __restrict__ bs2,
    float* __restrict__ out, int P)
{
    int wave = (blockIdx.x * 256 + threadIdx.x) >> 5;
    int lane = threadIdx.x & 31;
    if (wave >= P) return;
    const float* ha = h + (size_t)od[wave]     * HID;
    const float* hb = h + (size_t)od[P + wave] * HID;
    float partial = 0.f;
#pragma unroll
    for (int half = 0; half < 2; ++half) {
        int c = lane + half * 32;
        float acc = bs1[c];
        for (int k = 0; k < HID; ++k) acc += ha[k] * Ws1[k * HID + c];
        for (int k = 0; k < HID; ++k) acc += hb[k] * Ws1[(HID + k) * HID + c];
        partial += fmaxf(acc, 0.f) * Ws2[c];
    }
    for (int off = 16; off > 0; off >>= 1) partial += __shfl_down(partial, off, 32);
    if (lane == 0) out[wave] = partial + bs2[0];
}

// ---------------------------------------------------------------------------
// column sums of h -> ge[64]
// ---------------------------------------------------------------------------
__global__ void __launch_bounds__(256) colsum_kernel(
    const float* __restrict__ h, float* __restrict__ ge, int n)
{
    __shared__ float s[256];
    int c  = threadIdx.x & 63;
    int rg = blockIdx.x * 4 + (threadIdx.x >> 6);
    float acc = 0.f;
    for (int row = rg; row < n; row += gridDim.x * 4) acc += h[(size_t)row * HID + c];
    s[threadIdx.x] = acc;
    __syncthreads();
    if (threadIdx.x < 64) {
        float v = s[threadIdx.x] + s[threadIdx.x + 64] + s[threadIdx.x + 128] + s[threadIdx.x + 192];
        unsafeAtomicAdd(&ge[threadIdx.x], v);
    }
}

// ---------------------------------------------------------------------------
// dynamic-K head: x=[mean(h)|ts] -> 68->32->16->1 -> sigmoid scale
// ---------------------------------------------------------------------------
__global__ void __launch_bounds__(64) khead_kernel(
    const float* __restrict__ ge, const float* __restrict__ ts,
    const float* __restrict__ Wk1, const float* __restrict__ bk1,
    const float* __restrict__ Wk2, const float* __restrict__ bk2,
    const float* __restrict__ Wk3, const float* __restrict__ bk3,
    float* __restrict__ out, float invN)
{
    __shared__ float x[68];
    __shared__ float h1[32];
    __shared__ float h2[16];
    int t = threadIdx.x;
    if (t < 64) x[t] = ge[t] * invN;
    if (t < 4)  x[64 + t] = ts[t];
    __syncthreads();
    if (t < 32) {
        float acc = bk1[t];
        for (int k = 0; k < 68; ++k) acc += x[k] * Wk1[k * 32 + t];
        h1[t] = fmaxf(acc, 0.f);
    }
    __syncthreads();
    if (t < 16) {
        float acc = bk2[t];
        for (int k = 0; k < 32; ++k) acc += h1[k] * Wk2[k * 16 + t];
        h2[t] = fmaxf(acc, 0.f);
    }
    __syncthreads();
    if (t == 0) {
        float acc = bk3[0];
        for (int k = 0; k < 16; ++k) acc += h2[k] * Wk3[k];
        float sig = 1.f / (1.f + __expf(-acc));
        out[0] = 1.0f + sig * (50.0f - 1.0f);
    }
}

// ---------------------------------------------------------------------------
extern "C" void kernel_launch(void* const* d_in, const int* in_sizes, int n_in,
                              void* d_out, int out_size, void* d_ws, size_t ws_size,
                              hipStream_t stream) {
    const float* nf  = (const float*)d_in[0];
    const int*   ei  = (const int*)  d_in[1];
    const float* ef  = (const float*)d_in[2];
    const int*   od  = (const int*)  d_in[3];
    const float* ts  = (const float*)d_in[4];
    const float* Wn  = (const float*)d_in[5];
    const float* bn  = (const float*)d_in[6];
    const float* We  = (const float*)d_in[7];
    const float* be  = (const float*)d_in[8];
    const float* Wm  = (const float*)d_in[9];
    const float* bm  = (const float*)d_in[10];
    const float* Wu  = (const float*)d_in[11];
    const float* bu  = (const float*)d_in[12];
    const float* Ws1 = (const float*)d_in[13];
    const float* bs1 = (const float*)d_in[14];
    const float* Ws2 = (const float*)d_in[15];
    const float* bs2 = (const float*)d_in[16];
    const float* Wk1 = (const float*)d_in[17];
    const float* bk1 = (const float*)d_in[18];
    const float* Wk2 = (const float*)d_in[19];
    const float* bk2 = (const float*)d_in[20];
    const float* Wk3 = (const float*)d_in[21];
    const float* bk3 = (const float*)d_in[22];

    float* out = (float*)d_out;
    float* ws  = (float*)d_ws;
    float* h   = ws;                       // 3.2M f32
    float* agg = ws + (size_t)N_NODES*HID; // 3.2M f32
    float* ge  = agg + (size_t)N_NODES*HID;

    const int* srcp = ei;
    const int* dstp = ei + N_EDGES;

    nodeproj_kernel<<<(N_NODES * HID + 255) / 256, 256, 0, stream>>>(nf, Wn, bn, h, N_NODES);

    const int edgeGroups = N_EDGES / 128;          // 9375 (exact)
    const int nodeTiles  = N_NODES / 16;           // 3125 (exact)
    const int nodeGroups = (nodeTiles + 7) / 8;    // 391

    for (int l = 0; l < 3; ++l) {
        zero_kernel<<<512, 256, 0, stream>>>(agg, N_NODES * HID);
        msg_kernel<<<2048, 256, 0, stream>>>(h, ef, srcp, dstp, We, be,
                                             Wm + (size_t)l * 128 * HID, bm + (size_t)l * HID,
                                             agg, edgeGroups);
        upd_kernel<<<nodeGroups, 256, 0, stream>>>(h, agg,
                                                   Wu + (size_t)l * 128 * HID, bu + (size_t)l * HID,
                                                   nodeTiles, nodeGroups);
    }

    score_kernel<<<(N_OD * 32 + 255) / 256, 256, 0, stream>>>(h, od, Ws1, bs1, Ws2, bs2, out, N_OD);

    zero_kernel<<<1, 256, 0, stream>>>(ge, HID);
    colsum_kernel<<<128, 256, 0, stream>>>(h, ge, N_NODES);
    khead_kernel<<<1, 64, 0, stream>>>(ge, ts, Wk1, bk1, Wk2, bk2, Wk3, bk3,
                                       out + N_OD, 1.0f / (float)N_NODES);
}
